// SlidingKernelAttention2D_5136780886523
// MI455X (gfx1250) — matrix operand order repaired and checked by reading, not compiled, not run on hardware
//
#include <hip/hip_runtime.h>
#include <cstdint>
#include <cstddef>

typedef __attribute__((ext_vector_type(16))) _Float16 v16h;
typedef __attribute__((ext_vector_type(8)))  float    v8f;

#define DIMC  512
#define HEADS 8
#define DH    64
#define HH    64
#define WWID  64
#define NB    4
#define NWIN  63                 // H - KS + 1
#define NPIX  (NB*HH*WWID)       // 16384
#define BW    (NB*NWIN*NWIN)     // 15876

union FragAB { uint4 u[2]; v16h v; };
union H8     { uint4 u; _Float16 h[8]; };

// ---------------------------------------------------------------------------
// Kernel 1: NCHW f32 -> per-pixel row-major f16 [pixel][C] via LDS transpose
// grid.x = B*H (b,h), grid.y = C/16, block = 256
// ---------------------------------------------------------------------------
__global__ void k_pack_x(const float* __restrict__ x, _Float16* __restrict__ xh) {
  __shared__ __align__(16) _Float16 tile[WWID * 16]; // [w][c]
  const int bh = blockIdx.x;
  const int b  = bh >> 6;
  const int h  = bh & 63;
  const int c0 = blockIdx.y * 16;
  const int tid = threadIdx.x;
  for (int idx = tid; idx < 16 * WWID; idx += 256) {
    int c = idx >> 6;      // 0..15
    int w = idx & 63;
    tile[w * 16 + c] = (_Float16)x[(((size_t)b * DIMC + (c0 + c)) * HH + h) * WWID + w];
  }
  __syncthreads();
  int w    = tid >> 2;
  int part = tid & 3;
  size_t p = ((size_t)b * HH + h) * WWID + w;
  *(uint2*)(xh + p * DIMC + c0 + part * 4) = *(const uint2*)(tile + w * 16 + part * 4);
}

// ---------------------------------------------------------------------------
// Kernel 2: elementwise f32 -> f16
// ---------------------------------------------------------------------------
__global__ void k_f32_to_f16(const float* __restrict__ s, _Float16* __restrict__ d, int n) {
  int i = blockIdx.x * 256 + threadIdx.x;
  if (i < n) d[i] = (_Float16)s[i];
}

// ---------------------------------------------------------------------------
// WMMA fragment loaders (CDNA5 16x16x32 f16 layouts, ISA 7.12.2):
//   A: lane L holds row M=L&15; half h -> K = 16*(h>>3) + 8*(L>>4) + (h&7)
//   B: lane L holds col N=L&15; half h -> K = 16*(h>>3) + 8*(L>>4) + (h&7)   (B = W^T trick)
// ---------------------------------------------------------------------------
__device__ __forceinline__ v16h load_fragA(const _Float16* row, int k0, int hi) {
  FragAB a;
  const _Float16* ap = row + k0 + 8 * hi;
  a.u[0] = *(const uint4*)(ap);
  a.u[1] = *(const uint4*)(ap + 16);
  return a.v;
}
__device__ __forceinline__ v16h load_fragB(const _Float16* wrow, int k0, int hi) {
  FragAB b;
  const _Float16* bp = wrow + k0 + 8 * hi;
  b.u[0] = *(const uint4*)(bp);
  b.u[1] = *(const uint4*)(bp + 16);
  return b.v;
}

// ---------------------------------------------------------------------------
// Shared inner GEMM: one wave computes a 16x64 strip (4 accumulators).
// Software-pipelined: the K-step (k+32) fragment loads are issued in program
// order BEFORE the K-step (k) WMMAs, so each 4-WMMA burst hides the latency
// of the next step's 10 b128 loads instead of stalling on loadcnt==0.
// ---------------------------------------------------------------------------
__device__ __forceinline__ void gemm_strip(const _Float16* __restrict__ arow,
                                           const _Float16* __restrict__ wrow0,
                                           const _Float16* __restrict__ wrow1,
                                           const _Float16* __restrict__ wrow2,
                                           const _Float16* __restrict__ wrow3,
                                           int hi,
                                           v8f& acc0, v8f& acc1, v8f& acc2, v8f& acc3) {
  v16h a  = load_fragA(arow, 0, hi);
  v16h b0 = load_fragB(wrow0, 0, hi);
  v16h b1 = load_fragB(wrow1, 0, hi);
  v16h b2 = load_fragB(wrow2, 0, hi);
  v16h b3 = load_fragB(wrow3, 0, hi);
#pragma unroll
  for (int k0 = 0; k0 < DIMC; k0 += 32) {
    v16h ac = a, c0 = b0, c1 = b1, c2 = b2, c3 = b3;
    if (k0 + 32 < DIMC) {            // static after full unroll
      a  = load_fragA(arow, k0 + 32, hi);
      b0 = load_fragB(wrow0, k0 + 32, hi);
      b1 = load_fragB(wrow1, k0 + 32, hi);
      b2 = load_fragB(wrow2, k0 + 32, hi);
      b3 = load_fragB(wrow3, k0 + 32, hi);
    }
    acc0 = __builtin_amdgcn_wmma_f32_16x16x32_f16(false, ac, false, c0, (short)0, acc0, false, false);
    acc1 = __builtin_amdgcn_wmma_f32_16x16x32_f16(false, ac, false, c1, (short)0, acc1, false, false);
    acc2 = __builtin_amdgcn_wmma_f32_16x16x32_f16(false, ac, false, c2, (short)0, acc2, false, false);
    acc3 = __builtin_amdgcn_wmma_f32_16x16x32_f16(false, ac, false, c3, (short)0, acc3, false, false);
  }
}

// ---------------------------------------------------------------------------
// Kernel 3: qkv[p][n] = xh[p][:] . wq[n][:],  M=16384, N=1536, K=512, f16 out
// grid (M/16, N/256), block 128 (4 waves x 16x64 strip)
// ---------------------------------------------------------------------------
__global__ void k_gemm_qkv(const _Float16* __restrict__ xh,
                           const _Float16* __restrict__ wq,
                           _Float16* __restrict__ qkvh) {
  __shared__ __align__(16) _Float16 As[16 * DIMC];
  const int m0  = blockIdx.x * 16;
  const int tid = threadIdx.x;
  // A tile is a contiguous 16KB block (rows m0..m0+15 of xh)
  {
    const uint4* src = (const uint4*)(xh + (size_t)m0 * DIMC);
    uint4* dst = (uint4*)As;
    for (int idx = tid; idx < 16 * (DIMC / 8); idx += 128) dst[idx] = src[idx];
  }
  __syncthreads();
  const int lane  = tid & 31;
  const int hi    = lane >> 4;
  const int lr    = lane & 15;
  const int nbase = blockIdx.y * 256 + (tid >> 5) * 64 + lr;
  const _Float16* arow = As + lr * DIMC;
  v8f acc0 = {}, acc1 = {}, acc2 = {}, acc3 = {};
  gemm_strip(arow,
             wq + (size_t)(nbase)      * DIMC,
             wq + (size_t)(nbase + 16) * DIMC,
             wq + (size_t)(nbase + 32) * DIMC,
             wq + (size_t)(nbase + 48) * DIMC,
             hi, acc0, acc1, acc2, acc3);
  // D layout: element r of lane -> row M = r + 8*(lane>>4), col N = lane&15
#pragma unroll
  for (int r = 0; r < 8; ++r) {
    size_t mrow = (size_t)(m0 + r + 8 * hi) * (3 * DIMC);
    qkvh[mrow + nbase]      = (_Float16)acc0[r];
    qkvh[mrow + nbase + 16] = (_Float16)acc1[r];
    qkvh[mrow + nbase + 32] = (_Float16)acc2[r];
    qkvh[mrow + nbase + 48] = (_Float16)acc3[r];
  }
}

// ---------------------------------------------------------------------------
// Kernel 4: per-(window, head) 4x4 attention with decomposed rel-pos bias
// ---------------------------------------------------------------------------
__global__ void k_attn(const _Float16* __restrict__ qkvh,
                       const float* __restrict__ rel_h,
                       const float* __restrict__ rel_w,
                       _Float16* __restrict__ obuf) {
  int gid = blockIdx.x * 256 + threadIdx.x;
  if (gid >= BW * HEADS) return;
  const int head = gid & (HEADS - 1);
  const int win  = gid >> 3;
  const int j = win % NWIN;
  const int t = win / NWIN;
  const int i = t % NWIN;
  const int b = t / NWIN;

  const _Float16* tok[4];
#pragma unroll
  for (int l = 0; l < 4; ++l) {
    size_t p = ((size_t)b * HH + (i + (l >> 1))) * WWID + (j + (l & 1));
    tok[l] = qkvh + p * (3 * DIMC) + head * DH;
  }

  float dots[4][4] = {};
  float bh[4][2] = {};
  float bw[4][2] = {};

  for (int dc = 0; dc < DH; dc += 8) {
    float q8[4][8], k8[4][8];
#pragma unroll
    for (int l = 0; l < 4; ++l) {
      H8 qq, kk;
      qq.u = *(const uint4*)(tok[l] + dc);             // q
      kk.u = *(const uint4*)(tok[l] + DIMC + dc);      // k
#pragma unroll
      for (int d = 0; d < 8; ++d) { q8[l][d] = (float)qq.h[d]; k8[l][d] = (float)kk.h[d]; }
    }
    float rh[2][8], rw[2][8];
#pragma unroll
    for (int m = 0; m < 2; ++m) {
      float4 h0 = *(const float4*)(rel_h + m * DH + dc);
      float4 h1 = *(const float4*)(rel_h + m * DH + dc + 4);
      float4 w0 = *(const float4*)(rel_w + m * DH + dc);
      float4 w1 = *(const float4*)(rel_w + m * DH + dc + 4);
      rh[m][0] = h0.x; rh[m][1] = h0.y; rh[m][2] = h0.z; rh[m][3] = h0.w;
      rh[m][4] = h1.x; rh[m][5] = h1.y; rh[m][6] = h1.z; rh[m][7] = h1.w;
      rw[m][0] = w0.x; rw[m][1] = w0.y; rw[m][2] = w0.z; rw[m][3] = w0.w;
      rw[m][4] = w1.x; rw[m][5] = w1.y; rw[m][6] = w1.z; rw[m][7] = w1.w;
    }
#pragma unroll
    for (int l = 0; l < 4; ++l) {
#pragma unroll
      for (int lp = 0; lp < 4; ++lp)
#pragma unroll
        for (int d = 0; d < 8; ++d) dots[l][lp] += q8[l][d] * k8[lp][d];
#pragma unroll
      for (int m = 0; m < 2; ++m)
#pragma unroll
        for (int d = 0; d < 8; ++d) {
          bh[l][m] += q8[l][d] * rh[m][d];
          bw[l][m] += q8[l][d] * rw[m][d];
        }
    }
  }

  float attn[4][4];
#pragma unroll
  for (int l = 0; l < 4; ++l) {
    float mx = -3.0e38f;
#pragma unroll
    for (int lp = 0; lp < 4; ++lp) {
      float lg = dots[l][lp] * 0.125f + bh[l][lp >> 1] + bw[l][lp & 1]; // SCALE=1/8
      attn[l][lp] = lg;
      mx = fmaxf(mx, lg);
    }
    float sum = 0.f;
#pragma unroll
    for (int lp = 0; lp < 4; ++lp) { float e = __expf(attn[l][lp] - mx); attn[l][lp] = e; sum += e; }
    float inv = 1.0f / sum;
#pragma unroll
    for (int lp = 0; lp < 4; ++lp) attn[l][lp] *= inv;
  }

  for (int dc = 0; dc < DH; dc += 8) {
    float v8_[4][8];
#pragma unroll
    for (int lp = 0; lp < 4; ++lp) {
      H8 vv; vv.u = *(const uint4*)(tok[lp] + 2 * DIMC + dc);  // v
#pragma unroll
      for (int d = 0; d < 8; ++d) v8_[lp][d] = (float)vv.h[d];
    }
#pragma unroll
    for (int l = 0; l < 4; ++l) {
      H8 st;
#pragma unroll
      for (int d = 0; d < 8; ++d) {
        float o = attn[l][0] * v8_[0][d] + attn[l][1] * v8_[1][d] +
                  attn[l][2] * v8_[2][d] + attn[l][3] * v8_[3][d];
        st.h[d] = (_Float16)o;
      }
      *(uint4*)(obuf + ((size_t)win * 4 + l) * DIMC + head * DH + dc) = st.u;
    }
  }
}

// ---------------------------------------------------------------------------
// Kernel 5: per-pixel sum of <=4 overlapping window-token outputs (f16)
// ---------------------------------------------------------------------------
__global__ void k_gather(const _Float16* __restrict__ obuf, _Float16* __restrict__ acch) {
  int gid   = blockIdx.x * 256 + threadIdx.x;   // NPIX*64 threads exactly
  int chunk = gid & 63;
  int p     = gid >> 6;
  int w = p & 63, h = (p >> 6) & 63, b = p >> 12;
  float s[8] = {0, 0, 0, 0, 0, 0, 0, 0};
#pragma unroll
  for (int di = 0; di < 2; ++di)
#pragma unroll
    for (int dj = 0; dj < 2; ++dj) {
      int i = h - di, j = w - dj;
      if (i >= 0 && i < NWIN && j >= 0 && j < NWIN) {
        int win = (b * NWIN + i) * NWIN + j;
        int l = di * 2 + dj;
        H8 v; v.u = *(const uint4*)(obuf + ((size_t)win * 4 + l) * DIMC + chunk * 8);
#pragma unroll
        for (int d = 0; d < 8; ++d) s[d] += (float)v.h[d];
      }
    }
  H8 st;
#pragma unroll
  for (int d = 0; d < 8; ++d) st.h[d] = (_Float16)s[d];
  *(uint4*)(acch + (size_t)p * DIMC + chunk * 8) = st.u;
}

// ---------------------------------------------------------------------------
// Kernel 6: out = acc @ w_out^T + cnt(pixel)*b_out, scattered into NCHW f32
// grid (M/16, N/256), block 128; same pipelined 16x64 strip as kernel 3.
// ---------------------------------------------------------------------------
__global__ void k_gemm_out(const _Float16* __restrict__ acch,
                           const _Float16* __restrict__ wo,
                           const float* __restrict__ b_out,
                           float* __restrict__ out) {
  __shared__ __align__(16) _Float16 As[16 * DIMC];
  const int m0  = blockIdx.x * 16;
  const int tid = threadIdx.x;
  {
    const uint4* src = (const uint4*)(acch + (size_t)m0 * DIMC);
    uint4* dst = (uint4*)As;
    for (int idx = tid; idx < 16 * (DIMC / 8); idx += 128) dst[idx] = src[idx];
  }
  __syncthreads();
  const int lane  = tid & 31;
  const int hi    = lane >> 4;
  const int lr    = lane & 15;
  const int nbase = blockIdx.y * 256 + (tid >> 5) * 64 + lr;
  const _Float16* arow = As + lr * DIMC;
  v8f acc0 = {}, acc1 = {}, acc2 = {}, acc3 = {};
  gemm_strip(arow,
             wo + (size_t)(nbase)      * DIMC,
             wo + (size_t)(nbase + 16) * DIMC,
             wo + (size_t)(nbase + 32) * DIMC,
             wo + (size_t)(nbase + 48) * DIMC,
             hi, acc0, acc1, acc2, acc3);
  float b0 = b_out[nbase], b1 = b_out[nbase + 16], b2 = b_out[nbase + 32], b3 = b_out[nbase + 48];
#pragma unroll
  for (int r = 0; r < 8; ++r) {
    int m = m0 + r + 8 * hi;
    int w = m & 63, h = (m >> 6) & 63, b = m >> 12;
    float cnt = (float)(((h > 0) + (h < 63)) * ((w > 0) + (w < 63))); // windows covering pixel
    size_t base = (((size_t)b * DIMC) * HH + h) * WWID + w;           // + n*HH*WWID
    out[base + (size_t)(nbase)      * (HH * WWID)] = acc0[r] + cnt * b0;
    out[base + (size_t)(nbase + 16) * (HH * WWID)] = acc1[r] + cnt * b1;
    out[base + (size_t)(nbase + 32) * (HH * WWID)] = acc2[r] + cnt * b2;
    out[base + (size_t)(nbase + 48) * (HH * WWID)] = acc3[r] + cnt * b3;
  }
}

// ---------------------------------------------------------------------------
extern "C" void kernel_launch(void* const* d_in, const int* in_sizes, int n_in,
                              void* d_out, int out_size, void* d_ws, size_t ws_size,
                              hipStream_t stream) {
  const float* x     = (const float*)d_in[0];
  const float* w_qkv = (const float*)d_in[1];
  const float* w_out = (const float*)d_in[2];
  const float* b_out = (const float*)d_in[3];
  const float* rel_h = (const float*)d_in[4];
  const float* rel_w = (const float*)d_in[5];
  float* out = (float*)d_out;

  // workspace carve (256B-aligned offsets), total ~151 MB
  char* ws = (char*)d_ws;
  _Float16* xh   = (_Float16*)(ws);                 // 16384*512*2   = 16,777,216
  _Float16* wqh  = (_Float16*)(ws + 16777216);      // 1536*512*2    =  1,572,864
  _Float16* woh  = (_Float16*)(ws + 18350080);      // 512*512*2     =    524,288
  _Float16* qkvh = (_Float16*)(ws + 18874368);      // 16384*1536*2  = 50,331,648
  _Float16* obuf = (_Float16*)(ws + 69206016);      // 15876*4*512*2 = 65,028,096
  _Float16* acch = (_Float16*)(ws + 134234112);     // 16384*512*2   = 16,777,216

  k_pack_x<<<dim3(NB * HH, DIMC / 16), 256, 0, stream>>>(x, xh);
  k_f32_to_f16<<<(3 * DIMC * DIMC) / 256, 256, 0, stream>>>(w_qkv, wqh, 3 * DIMC * DIMC);
  k_f32_to_f16<<<(DIMC * DIMC) / 256, 256, 0, stream>>>(w_out, woh, DIMC * DIMC);
  k_gemm_qkv<<<dim3(NPIX / 16, (3 * DIMC) / 256), 128, 0, stream>>>(xh, wqh, qkvh);
  k_attn<<<(BW * HEADS + 255) / 256, 256, 0, stream>>>(qkvh, rel_h, rel_w, obuf);
  k_gather<<<(NPIX * 64) / 256, 256, 0, stream>>>(obuf, acch);
  k_gemm_out<<<dim3(NPIX / 16, DIMC / 256), 128, 0, stream>>>(acch, woh, b_out, out);
}
